// SelfAttention2d_9869834846799
// MI455X (gfx1250) — matrix operand order repaired and checked by reading, not compiled, not run on hardware
//
#include <hip/hip_runtime.h>
#include <hip/hip_bf16.h>

// ---------------------------------------------------------------------------
// SelfAttention2d on gfx1250 (MI455X): bf16 WMMA GEMMs + fp32 softmax.
// B=8, C=512, S=H*W=1024, heads=8, dh=64.
// All matrix math via v_wmma_f32_16x16x32_bf16. Attention score tile lives in
// LDS (never written to HBM). Q tile staged to LDS via async-to-LDS (ASYNCcnt)
// when the toolchain exposes the builtin. Requires ~44MB of d_ws.
// ---------------------------------------------------------------------------

typedef __attribute__((ext_vector_type(16))) __bf16 v16bf;
typedef __attribute__((ext_vector_type(8)))  float  v8f;
typedef __attribute__((ext_vector_type(4)))  int    v4i;

#define AS_GLOBAL __attribute__((address_space(1)))
#define AS_LDS    __attribute__((address_space(3)))

union Frag16 {
    v16bf v;
    uint4 q[2];
    __bf16 e[16];
};

union Pack8 {
    __bf16 e[8];
    uint4 q;
};

// A-fragment (16x32, M x K), row-major source with leading dim ld.
// Lane layout: row = lane&15, K = k0 + (lane>>4)*16 .. +15 (16 contiguous),
//              matching load_bt_frag.
__device__ __forceinline__ Frag16 load_a_frag(const __bf16* __restrict__ base, int ld) {
    int lane = threadIdx.x & 31;
    int m  = lane & 15;
    int kb = (lane >> 4) * 16;
    const __bf16* p = base + (size_t)m * ld + kb;
    Frag16 f;
    f.q[0] = *(const uint4*)(p);        // K = kb .. kb+7
    f.q[1] = *(const uint4*)(p + 8);    // K = kb+8 .. kb+15
    return f;
}

// B-fragment (32x16, K x N) loaded from B^T stored row-major [N][K] (ld = K stride).
// Lane layout: col n = lane&15, K = k0 + (lane>>4)*16 .. +15 (16 contiguous).
__device__ __forceinline__ Frag16 load_bt_frag(const __bf16* __restrict__ baseT, int ld) {
    int lane = threadIdx.x & 31;
    int n  = lane & 15;
    int kb = (lane >> 4) * 16;
    const __bf16* p = baseT + (size_t)n * ld + kb;
    Frag16 f;
    f.q[0] = *(const uint4*)(p);
    f.q[1] = *(const uint4*)(p + 8);
    return f;
}

__device__ __forceinline__ v8f wmma_bf16(const Frag16& a, const Frag16& b, v8f c) {
    return __builtin_amdgcn_wmma_f32_16x16x32_bf16(
        /*neg_a=*/false, a.v, /*neg_b=*/false, b.v,
        /*c_mod=*/(short)0, c, /*reuse_a=*/false, /*reuse_b=*/false);
}

__device__ __forceinline__ v8f zero_v8f() {
    v8f z = {0.f, 0.f, 0.f, 0.f, 0.f, 0.f, 0.f, 0.f};
    return z;
}

// ---------------------------------------------------------------------------
// Kernel 0a: x [8][512][1024] f32  ->  xt [8][1024][512] bf16 (transposed)
// ---------------------------------------------------------------------------
__global__ void sa2d_cvt_x(const float* __restrict__ x, __bf16* __restrict__ xt) {
    int idx = blockIdx.x * blockDim.x + threadIdx.x;     // 8*512*1024 = 2^22
    int s = idx & 1023;
    int c = (idx >> 10) & 511;
    int b = idx >> 19;
    xt[((size_t)b * 1024 + s) * 512 + c] = (__bf16)x[idx];
}

// Kernel 0b: weight f32 -> bf16
__global__ void sa2d_cvt_w(const float* __restrict__ w, __bf16* __restrict__ wb, int n) {
    int i = blockIdx.x * blockDim.x + threadIdx.x;
    if (i < n) wb[i] = (__bf16)w[i];
}

// ---------------------------------------------------------------------------
// Kernel 1: fused QKV projection.
// Per batch b: M=512 (out ch), N=1024 (s), K=512.  A = W [o][c] row-major,
// B^T = xt[b][s][c].  Each wave: one 16x64 strip (A-frag reused 4x).
// Q,K stored transposed qt/kt[b][s][c] (vector stores); V natural [b][c][s].
// grid = (16 n-groups, 4 m-groups, b*3+which), block = 256 (8 waves)
// ---------------------------------------------------------------------------
__global__ void sa2d_proj_qkv(const __bf16* __restrict__ xt,
                              const __bf16* __restrict__ wq,
                              const __bf16* __restrict__ wk,
                              const __bf16* __restrict__ wv,
                              const float*  __restrict__ bq,
                              const float*  __restrict__ bk,
                              const float*  __restrict__ bv,
                              __bf16* __restrict__ qt,
                              __bf16* __restrict__ kt,
                              __bf16* __restrict__ vmat) {
    const int C = 512, S = 1024;
    int wave = threadIdx.x >> 5;
    int lane = threadIdx.x & 31;
    int n0 = blockIdx.x * 64;
    int m0 = blockIdx.y * 128 + wave * 16;
    int b  = blockIdx.z / 3;
    int which = blockIdx.z % 3;

    const __bf16* W    = (which == 0) ? wq : (which == 1) ? wk : wv;
    const float*  bias = (which == 0) ? bq : (which == 1) ? bk : bv;
    const __bf16* xb   = xt + (size_t)b * S * C;

    v8f acc[4] = {zero_v8f(), zero_v8f(), zero_v8f(), zero_v8f()};

    for (int k0 = 0; k0 < C; k0 += 32) {
        if (k0 + 32 < C) {  // prefetch next K-slice (lowers to global_prefetch_b8)
            __builtin_prefetch(W + (size_t)(m0 + (lane & 15)) * C + k0 + 32, 0, 1);
            __builtin_prefetch(xb + (size_t)(n0 + (lane & 15)) * C + k0 + 32, 0, 1);
        }
        Frag16 a = load_a_frag(W + (size_t)m0 * C + k0, C);
#pragma unroll
        for (int j = 0; j < 4; ++j) {
            Frag16 bf = load_bt_frag(xb + (size_t)(n0 + j * 16) * C + k0, C);
            acc[j] = wmma_bf16(a, bf, acc[j]);
        }
    }

    int rbase = (lane >> 4) * 8;   // row offset within tile for this lane's 8 values
    int ncol  = lane & 15;
    if (which < 2) {
        // transposed store: out[s][c], 8 contiguous bf16 per lane
        __bf16* out = ((which == 0) ? qt : kt) + (size_t)b * S * C;
#pragma unroll
        for (int j = 0; j < 4; ++j) {
            int n = n0 + j * 16 + ncol;
            Pack8 p;
#pragma unroll
            for (int r = 0; r < 8; ++r)
                p.e[r] = (__bf16)(acc[j][r] + bias[m0 + rbase + r]);
            *(uint4*)(out + (size_t)n * C + m0 + rbase) = p.q;
        }
    } else {
        // natural store: vmat[c][s]
        __bf16* out = vmat + (size_t)b * C * S;
#pragma unroll
        for (int j = 0; j < 4; ++j) {
            int n = n0 + j * 16 + ncol;
#pragma unroll
            for (int r = 0; r < 8; ++r)
                out[(size_t)(m0 + rbase + r) * S + n] =
                    (__bf16)(acc[j][r] + bias[m0 + rbase + r]);
        }
    }
}

// ---------------------------------------------------------------------------
// Kernel 2: attention per (b,g): scores (LDS f32) -> softmax -> O = V*P^T.
// grid = (32 s-tiles of 32 rows, 64 bg), block = 256 (8 waves)
// dynamic LDS: 32*1024*4 (scores) + 32*1024*2 (probs bf16) + 32*64*2 (Q tile)
//            = 200704 B
// ---------------------------------------------------------------------------
__global__ void sa2d_attn(const __bf16* __restrict__ qt,
                          const __bf16* __restrict__ kt,
                          const __bf16* __restrict__ vmat,
                          __bf16* __restrict__ ot) {
    const int C = 512, S = 1024, DH = 64;
    extern __shared__ char smem[];
    float*  sc = (float*)smem;                               // [32][1024] f32
    __bf16* P  = (__bf16*)(smem + 32 * 1024 * 4);            // [32][1024] bf16
    __bf16* qs = (__bf16*)(smem + 32 * 1024 * 4 + 32 * 1024 * 2); // [32][64] bf16

    int wave = threadIdx.x >> 5;
    int lane = threadIdx.x & 31;
    int s0 = blockIdx.x * 32;
    int bg = blockIdx.y;
    int b = bg >> 3, g = bg & 7;

    const __bf16* qb = qt + (size_t)b * S * C + g * DH;   // rows: s, cols: head ch
    const __bf16* kb = kt + (size_t)b * S * C + g * DH;
    const float scale = 0.044194173824159216f;            // 1/sqrt(512) (full dim!)

    // ---- stage Q tile [32 s][64 c] into LDS (4 KB), async when available ---
    {
        // 256 threads, each moves 8 bf16 (16B): 32 rows * 8 chunks
        int row = threadIdx.x >> 3;       // 0..31
        int c8  = (threadIdx.x & 7) * 8;  // 0,8,...,56
        const __bf16* gp = qb + (size_t)(s0 + row) * C + c8;
        __bf16*       lp = qs + row * DH + c8;
#if __has_builtin(__builtin_amdgcn_global_load_async_to_lds_b128)
        __builtin_amdgcn_global_load_async_to_lds_b128(
            (AS_GLOBAL v4i*)gp, (AS_LDS v4i*)lp, 0, 0);
#if __has_builtin(__builtin_amdgcn_s_wait_asynccnt)
        __builtin_amdgcn_s_wait_asynccnt(0);
#endif
#else
        *(uint4*)lp = *(const uint4*)gp;
#endif
    }
    __syncthreads();

    // ---- phase 1: scores tile [32 s][1024 t] into LDS ----------------------
    // 2 m-tiles x 64 n-tiles = 128 tiles; each wave does 16.
    for (int i = 0; i < 16; ++i) {
        int t  = wave * 16 + i;
        int tm = t >> 6;     // 0..1
        int tn = t & 63;     // 0..63
        v8f acc = zero_v8f();
        for (int k0 = 0; k0 < DH; k0 += 32) {
            Frag16 a  = load_a_frag(qs + (size_t)(tm * 16) * DH + k0, DH);  // from LDS
            Frag16 bf = load_bt_frag(kb + (size_t)(tn * 16) * C + k0, C);
            acc = wmma_bf16(a, bf, acc);
        }
        int m = tm * 16 + (lane >> 4) * 8;
        int n = tn * 16 + (lane & 15);
#pragma unroll
        for (int r = 0; r < 8; ++r)
            sc[(size_t)(m + r) * 1024 + n] = acc[r] * scale;
    }
    __syncthreads();

    // ---- phase 2: row softmax (fp32), requantize to bf16 -------------------
    for (int rr = 0; rr < 4; ++rr) {
        int row = wave + rr * 8;
        float* rp = sc + (size_t)row * 1024;
        float vals[32];
        float mx = -3.0e38f;
#pragma unroll
        for (int i = 0; i < 32; ++i) {
            vals[i] = rp[lane + 32 * i];
            mx = fmaxf(mx, vals[i]);
        }
#pragma unroll
        for (int off = 16; off > 0; off >>= 1)
            mx = fmaxf(mx, __shfl_xor(mx, off, 32));
        float sum = 0.f;
#pragma unroll
        for (int i = 0; i < 32; ++i) {
            vals[i] = __expf(vals[i] - mx);
            sum += vals[i];
        }
#pragma unroll
        for (int off = 16; off > 0; off >>= 1)
            sum += __shfl_xor(sum, off, 32);
        float inv = 1.0f / sum;
        __bf16* pp = P + (size_t)row * 1024;
#pragma unroll
        for (int i = 0; i < 32; ++i)
            pp[lane + 32 * i] = (__bf16)(vals[i] * inv);
    }
    __syncthreads();

    // ---- phase 3: O[c][s] = sum_t V[c][t] * P[s][t]  (M=64,N=32,K=1024) ----
    // 4 m-tiles x 2 n-tiles = 8 tiles; one per wave.
    int tm = wave >> 1;   // 0..3
    int tn = wave & 1;    // 0..1
    const __bf16* vb = vmat + ((size_t)b * C + g * DH) * S;
    v8f acc = zero_v8f();
    for (int k0 = 0; k0 < S; k0 += 32) {
        if (k0 + 32 < S)
            __builtin_prefetch(vb + (size_t)(tm * 16 + (lane & 15)) * S + k0 + 32, 0, 1);
        Frag16 a = load_a_frag(vb + (size_t)(tm * 16) * S + k0, S);
        Frag16 bf;
        {
            int n   = lane & 15;
            int kb2 = k0 + (lane >> 4) * 16;
            const uint4* p = (const uint4*)(P + (size_t)(tn * 16 + n) * 1024 + kb2);
            bf.q[0] = p[0];
            bf.q[1] = p[1];
        }
        acc = wmma_bf16(a, bf, acc);
    }
    // transposed store into ot[b][s][c]
    int ncol  = lane & 15;
    int rbase = (lane >> 4) * 8;
    int sgl = s0 + tn * 16 + ncol;
    Pack8 pk;
#pragma unroll
    for (int r = 0; r < 8; ++r) pk.e[r] = (__bf16)acc[r];
    *(uint4*)(ot + ((size_t)b * S + sgl) * C + g * DH + tm * 16 + rbase) = pk.q;
}

// ---------------------------------------------------------------------------
// Kernel 3: out = x + Wo * O + bo   (fp32 output, natural [b][c][s] layout)
// grid = (16, 4, 8), block = 256
// ---------------------------------------------------------------------------
__global__ void sa2d_proj_out(const __bf16* __restrict__ ot,
                              const __bf16* __restrict__ wo,
                              const float*  __restrict__ bo,
                              const float*  __restrict__ x,
                              float* __restrict__ out) {
    const int C = 512, S = 1024;
    int wave = threadIdx.x >> 5;
    int lane = threadIdx.x & 31;
    int n0 = blockIdx.x * 64;
    int m0 = blockIdx.y * 128 + wave * 16;
    int b  = blockIdx.z;

    const __bf16* ob = ot + (size_t)b * S * C;
    v8f acc[4] = {zero_v8f(), zero_v8f(), zero_v8f(), zero_v8f()};

    for (int k0 = 0; k0 < C; k0 += 32) {
        if (k0 + 32 < C) {
            __builtin_prefetch(wo + (size_t)(m0 + (lane & 15)) * C + k0 + 32, 0, 1);
            __builtin_prefetch(ob + (size_t)(n0 + (lane & 15)) * C + k0 + 32, 0, 1);
        }
        Frag16 a = load_a_frag(wo + (size_t)m0 * C + k0, C);
#pragma unroll
        for (int j = 0; j < 4; ++j) {
            Frag16 bf = load_bt_frag(ob + (size_t)(n0 + j * 16) * C + k0, C);
            acc[j] = wmma_bf16(a, bf, acc[j]);
        }
    }

    const float* xb   = x   + (size_t)b * C * S;
    float*       outb = out + (size_t)b * C * S;
    int rbase = (lane >> 4) * 8;
    int ncol  = lane & 15;
#pragma unroll
    for (int j = 0; j < 4; ++j) {
        int n = n0 + j * 16 + ncol;
#pragma unroll
        for (int r = 0; r < 8; ++r) {
            int m = m0 + rbase + r;
            outb[(size_t)m * S + n] = xb[(size_t)m * S + n] + bo[m] + acc[j][r];
        }
    }
}

// ---------------------------------------------------------------------------
extern "C" void kernel_launch(void* const* d_in, const int* in_sizes, int n_in,
                              void* d_out, int out_size, void* d_ws, size_t ws_size,
                              hipStream_t stream) {
    (void)in_sizes; (void)n_in; (void)out_size; (void)ws_size;

    const float* x  = (const float*)d_in[0];
    const float* Wq = (const float*)d_in[1];
    const float* bq = (const float*)d_in[2];
    const float* Wk = (const float*)d_in[3];
    const float* bk = (const float*)d_in[4];
    const float* Wv = (const float*)d_in[5];
    const float* bv = (const float*)d_in[6];
    const float* Wo = (const float*)d_in[7];
    const float* bo = (const float*)d_in[8];
    float* out = (float*)d_out;

    const size_t BSC = (size_t)8 * 1024 * 512;   // elements of each bf16 tensor
    const size_t WSZ = (size_t)512 * 512;
    const int    LDS_BYTES = 32 * 1024 * 4 + 32 * 1024 * 2 + 32 * 64 * 2; // 200704

    char* ws = (char*)d_ws;                      // needs ~44 MB
    __bf16* xt   = (__bf16*)ws;            ws += BSC * 2;
    __bf16* qt   = (__bf16*)ws;            ws += BSC * 2;
    __bf16* kt   = (__bf16*)ws;            ws += BSC * 2;
    __bf16* vmat = (__bf16*)ws;            ws += BSC * 2;
    __bf16* ot   = (__bf16*)ws;            ws += BSC * 2;
    __bf16* wqb  = (__bf16*)ws;            ws += WSZ * 2;
    __bf16* wkb  = (__bf16*)ws;            ws += WSZ * 2;
    __bf16* wvb  = (__bf16*)ws;            ws += WSZ * 2;
    __bf16* wob  = (__bf16*)ws;            ws += WSZ * 2;

    // allow ~196 KB dynamic LDS for the attention kernel
    (void)hipFuncSetAttribute((const void*)sa2d_attn,
                              hipFuncAttributeMaxDynamicSharedMemorySize, LDS_BYTES);

    // 0) precision conversion / layout
    sa2d_cvt_x<<<dim3(16384), dim3(256), 0, stream>>>(x, xt);
    sa2d_cvt_w<<<dim3(1024), dim3(256), 0, stream>>>(Wq, wqb, (int)WSZ);
    sa2d_cvt_w<<<dim3(1024), dim3(256), 0, stream>>>(Wk, wkb, (int)WSZ);
    sa2d_cvt_w<<<dim3(1024), dim3(256), 0, stream>>>(Wv, wvb, (int)WSZ);
    sa2d_cvt_w<<<dim3(1024), dim3(256), 0, stream>>>(Wo, wob, (int)WSZ);

    // 1) fused QKV projection
    sa2d_proj_qkv<<<dim3(16, 4, 24), dim3(256), 0, stream>>>(
        xt, wqb, wkb, wvb, bq, bk, bv, qt, kt, vmat);

    // 2) attention (scores + softmax + O)
    sa2d_attn<<<dim3(32, 64), dim3(256), LDS_BYTES, stream>>>(qt, kt, vmat, ot);

    // 3) output projection + residual
    sa2d_proj_out<<<dim3(16, 4, 8), dim3(256), 0, stream>>>(ot, wob, bo, x, out);
}
